// EncodeProcessDecodeModel_26929444945967
// MI455X (gfx1250) — matrix operand order repaired and checked by reading, not compiled, not run on hardware
//
#include <hip/hip_runtime.h>
#include <hip/hip_bf16.h>

typedef __attribute__((ext_vector_type(16))) _Float16 v16h;
typedef __attribute__((ext_vector_type(8)))  _Float16 v8h;
typedef __attribute__((ext_vector_type(8)))  float    v8f;
typedef __attribute__((ext_vector_type(4)))  float    v4f;
typedef __attribute__((ext_vector_type(4)))  int      v4i;

#define N_DATA 131072
#define N_HID  32768
#define DIM    256
#define E_CNT  524288
#define PITCH  40          // halves per LDS row: 32 data + 8 pad (80B pitch)
#define SLAB   (256 * PITCH)

#if __has_builtin(__builtin_amdgcn_global_load_async_to_lds_b128) && \
    __has_builtin(__builtin_amdgcn_s_wait_asynccnt)
#define HAVE_ASYNC_LDS 1
#else
#define HAVE_ASYNC_LDS 0
#endif

__device__ __forceinline__ float gelu_erf(float x) {
    return 0.5f * x * (1.0f + erff(x * 0.70710678118654752f));
}

// ---------------------------------------------------------------------------
// f32 -> f16 weight conversion (weights: 128KB each, L2 resident)
// ---------------------------------------------------------------------------
__global__ void cvt_f16_kernel(const float* __restrict__ in, _Float16* __restrict__ out, int n) {
    int i = blockIdx.x * blockDim.x + threadIdx.x;
    if (i < n) out[i] = (_Float16)in[i];
}

// ---------------------------------------------------------------------------
// Small-K encoder: out[node][n] = gelu( sum_k X[node][k]*W[n][k] + b[n] )
// ---------------------------------------------------------------------------
template<int K>
__global__ void enc_small_kernel(const float* __restrict__ X, const float* __restrict__ W,
                                 const float* __restrict__ b, float* __restrict__ out,
                                 int n_nodes) {
    int idx = blockIdx.x * blockDim.x + threadIdx.x;       // node*256 + n
    if (idx >= n_nodes * DIM) return;
    int node = idx >> 8;
    int n    = idx & 255;
    float s = b[n];
    const float* xr = X + (size_t)node * K;
    const float* wr = W + (size_t)n * K;
#pragma unroll
    for (int k = 0; k < K; ++k) s += xr[k] * wr[k];
    out[idx] = gelu_erf(s);
}

// ---------------------------------------------------------------------------
// Edge scatter-add (mean numerator + count). One wave per edge, 8 floats/lane.
// Aggregation buffers fit in 192MB L2, so atomics are L2-resident RMW.
// ---------------------------------------------------------------------------
__global__ void scatter_add_kernel(const float* __restrict__ src_feat,
                                   const int* __restrict__ src, const int* __restrict__ dst,
                                   float* __restrict__ agg, float* __restrict__ cnt, int E) {
    int t = blockIdx.x * blockDim.x + threadIdx.x;
    int e = t >> 5;
    int lane = t & 31;
    if (e >= E) return;
    int s = src[e];
    int d = dst[e];
    const float* sp = src_feat + (size_t)s * DIM + lane * 8;
    float*       dp = agg      + (size_t)d * DIM + lane * 8;
    v4f v0 = *(const v4f*)(sp);
    v4f v1 = *(const v4f*)(sp + 4);
#pragma unroll
    for (int i = 0; i < 4; ++i) {
        __hip_atomic_fetch_add(dp + i,     v0[i], __ATOMIC_RELAXED, __HIP_MEMORY_SCOPE_AGENT);
        __hip_atomic_fetch_add(dp + 4 + i, v1[i], __ATOMIC_RELAXED, __HIP_MEMORY_SCOPE_AGENT);
    }
    if (lane == 0)
        __hip_atomic_fetch_add(cnt + d, 1.0f, __ATOMIC_RELAXED, __HIP_MEMORY_SCOPE_AGENT);
}

// ---------------------------------------------------------------------------
// B fragment load from LDS slab (two 16B ds loads)
// ---------------------------------------------------------------------------
__device__ __forceinline__ v16h load_bfrag(const _Float16* sbase, int off) {
    const _Float16* bs = sbase + off;
    v8h b0 = *(const v8h*)(bs);
    v8h b1 = *(const v8h*)(bs + 16);
    v16h b;
#pragma unroll
    for (int i = 0; i < 8; ++i) { b[i] = b0[i]; b[8 + i] = b1[i]; }
    return b;
}

#if HAVE_ASYNC_LDS
// Issue 4 async 16B global->LDS copies for one 64B weight row slab.
// Builtin signature: (v4i addrspace(1)* src, v4i addrspace(3)* dst,
//                     imm int offset, imm int cpol); INST_OFFSET is added to
// both the global and LDS addresses, so one base pair + 4 offsets suffices.
__device__ __forceinline__ void async_stage_row(const _Float16* wp, _Float16* dst) {
    auto* g = (__attribute__((address_space(1))) v4i*)(wp);
    auto* l = (__attribute__((address_space(3))) v4i*)(dst);
    __builtin_amdgcn_global_load_async_to_lds_b128(g, l, 0,  0);
    __builtin_amdgcn_global_load_async_to_lds_b128(g, l, 16, 0);
    __builtin_amdgcn_global_load_async_to_lds_b128(g, l, 32, 0);
    __builtin_amdgcn_global_load_async_to_lds_b128(g, l, 48, 0);
}
#endif

// ---------------------------------------------------------------------------
// WMMA GEMM:  C[M,256] (+)= A[M,256] @ W^T   (W: [256 out][256 in] f16)
//   ACCUM:    C += result             !ACCUM: C = result + bias
//   GELU_IN:  gelu(A) fused into f32->f16 conversion
//   GELU_OUT: gelu in epilogue
//   ALIAS:    A may alias C (block owns its 64 rows; barrier before writeback)
//   MEAN_IN:  A row scaled by 1/max(cnt[m],1) during conversion (fused mean)
// Block: 256 thr = 8 waves; block tile 64x256; wave tile 16x128 (8 acc).
// Weight K-slab [256n x 32k] double-buffered in LDS (80B pitch); staged via
// async global->LDS copies when available, else reg round-trip.
// ---------------------------------------------------------------------------
template<bool ACCUM, bool GELU_IN, bool GELU_OUT, bool ALIAS, bool MEAN_IN>
__global__ __launch_bounds__(256)
void gemm256_wmma_kernel(const float* A, const _Float16* __restrict__ W,
                         const float* __restrict__ bias, float* C,
                         const float* __restrict__ cnt, int M) {
    __shared__ _Float16 smem[2 * SLAB];    // 2 x 20KB

    const int tid      = threadIdx.x;
    const int lane     = tid & 31;
    const int wv       = tid >> 5;
    const int m0       = blockIdx.x * 64 + (wv >> 1) * 16;  // 4 m-tiles
    const int n0       = (wv & 1) * 128;                    // 2 n-halves
    const int half_sel = lane >> 4;
    const int lr       = lane & 15;

    const float* arow = A + (size_t)(m0 + lr) * DIM;
    float scale = 1.0f;
    if (MEAN_IN) scale = 1.0f / fmaxf(cnt[m0 + lr], 1.0f);

    v8f acc[8];
#pragma unroll
    for (int j = 0; j < 8; ++j) acc[j] = (v8f){0.f,0.f,0.f,0.f,0.f,0.f,0.f,0.f};

#if HAVE_ASYNC_LDS
    // ---- prologue: async-stage slab 0 into buf0
    async_stage_row(W + (size_t)tid * DIM, smem + tid * PITCH);
#else
    // ---- prologue: stage slab 0 into buf0 via regs, prefetch slab 1
    v8h g0, g1, g2, g3;
    {
        const _Float16* wp = W + (size_t)tid * DIM;
        g0 = *(const v8h*)(wp);      g1 = *(const v8h*)(wp + 8);
        g2 = *(const v8h*)(wp + 16); g3 = *(const v8h*)(wp + 24);
        _Float16* s = smem + tid * PITCH;
        *(v8h*)(s) = g0; *(v8h*)(s + 8) = g1; *(v8h*)(s + 16) = g2; *(v8h*)(s + 24) = g3;
        const _Float16* wp1 = wp + 32;
        g0 = *(const v8h*)(wp1);      g1 = *(const v8h*)(wp1 + 8);
        g2 = *(const v8h*)(wp1 + 16); g3 = *(const v8h*)(wp1 + 24);
    }
#endif
    // ---- prefetch A fragment for kk=0
    const float* ap = arow + half_sel * 8;
    v4f f0 = *(const v4f*)(ap);
    v4f f1 = *(const v4f*)(ap + 4);
    v4f f2 = *(const v4f*)(ap + 16);
    v4f f3 = *(const v4f*)(ap + 20);

#pragma unroll
    for (int t = 0; t < 8; ++t) {
#if HAVE_ASYNC_LDS
        __builtin_amdgcn_s_wait_asynccnt(0);                // slab t landed in LDS
        __syncthreads();
        if (t + 1 < 8)                                      // async-stage slab t+1
            async_stage_row(W + (size_t)tid * DIM + (t + 1) * 32,
                            smem + ((t + 1) & 1) * SLAB + tid * PITCH);
#else
        __syncthreads();                                    // slab t visible in buf t&1
        if (t + 1 < 8) {
            _Float16* s = smem + ((t + 1) & 1) * SLAB + tid * PITCH;
            *(v8h*)(s) = g0; *(v8h*)(s + 8) = g1; *(v8h*)(s + 16) = g2; *(v8h*)(s + 24) = g3;
            if (t + 2 < 8) {
                const _Float16* wp = W + (size_t)tid * DIM + (t + 2) * 32;
                g0 = *(const v8h*)(wp);      g1 = *(const v8h*)(wp + 8);
                g2 = *(const v8h*)(wp + 16); g3 = *(const v8h*)(wp + 24);
            }
        }
#endif
        // ---- prefetch next A fragment
        v4f nf0, nf1, nf2, nf3;
        if (t + 1 < 8) {
            const float* apn = arow + (t + 1) * 32 + half_sel * 8;
            nf0 = *(const v4f*)(apn);
            nf1 = *(const v4f*)(apn + 4);
            nf2 = *(const v4f*)(apn + 16);
            nf3 = *(const v4f*)(apn + 20);
        }
        // ---- convert current A fragment (fused mean / gelu)
        v16h a;
#pragma unroll
        for (int i = 0; i < 4; ++i) {
            float e0 = f0[i], e1 = f1[i], e2 = f2[i], e3 = f3[i];
            if (MEAN_IN) { e0 *= scale; e1 *= scale; e2 *= scale; e3 *= scale; }
            if (GELU_IN) { e0 = gelu_erf(e0); e1 = gelu_erf(e1);
                           e2 = gelu_erf(e2); e3 = gelu_erf(e3); }
            a[i] = (_Float16)e0; a[4 + i] = (_Float16)e1;
            a[8 + i] = (_Float16)e2; a[12 + i] = (_Float16)e3;
        }
        // ---- 8 B fragments from LDS + 8 WMMAs, software-pipelined so the
        //      ds_load pair for j+1 is in flight while WMMA j executes.
        const _Float16* sbase = smem + (t & 1) * SLAB;
        v16h bcur = load_bfrag(sbase, (n0 + lr) * PITCH + half_sel * 8);
#pragma unroll
        for (int j = 0; j < 8; ++j) {
            v16h bnext;
            if (j + 1 < 8)
                bnext = load_bfrag(sbase, (n0 + 16 * (j + 1) + lr) * PITCH + half_sel * 8);
            acc[j] = __builtin_amdgcn_wmma_f32_16x16x32_f16(
                false, a, false, bcur, (short)0, acc[j], false, false);
            bcur = bnext;
        }
        f0 = nf0; f1 = nf1; f2 = nf2; f3 = nf3;
    }

    if (ALIAS) __syncthreads();   // all reads of block-owned A rows before writes

    // ---- epilogue. C/D layout: VGPR i -> row (i + 8*half_sel), col lr.
#pragma unroll
    for (int j = 0; j < 8; ++j) {
        int n = n0 + 16 * j + lr;
#pragma unroll
        for (int i = 0; i < 8; ++i) {
            int m = m0 + i + 8 * half_sel;
            size_t idx = (size_t)m * DIM + n;
            float v = acc[j][i];
            if (ACCUM) v += C[idx];
            else       v += bias[n];
            if (GELU_OUT) v = gelu_erf(v);
            C[idx] = v;
        }
    }
}

// ---------------------------------------------------------------------------
// Output projection: out[node][o] = sum_k D[node][k]*Wo[o][k] + bo[o], o<8
// ---------------------------------------------------------------------------
__global__ void out_proj_kernel(const float* __restrict__ D, const float* __restrict__ Wo,
                                const float* __restrict__ bo, float* __restrict__ out,
                                int n_nodes) {
    int t = blockIdx.x * blockDim.x + threadIdx.x;    // node*8 + o
    if (t >= n_nodes * 8) return;
    int node = t >> 3;
    int o    = t & 7;
    const float* dr = D  + (size_t)node * DIM;
    const float* wr = Wo + (size_t)o * DIM;
    float s = bo[o];
#pragma unroll 4
    for (int k = 0; k < DIM; k += 4) {
        v4f a = *(const v4f*)(dr + k);
        v4f b = *(const v4f*)(wr + k);
        s += a[0]*b[0] + a[1]*b[1] + a[2]*b[2] + a[3]*b[3];
    }
    out[t] = s;
}

// ---------------------------------------------------------------------------
extern "C" void kernel_launch(void* const* d_in, const int* in_sizes, int n_in,
                              void* d_out, int out_size, void* d_ws, size_t ws_size,
                              hipStream_t stream) {
    const float* data_x    = (const float*)d_in[0];
    const float* hidden_x  = (const float*)d_in[1];
    const int*   dh_src    = (const int*)d_in[2];
    const int*   dh_dst    = (const int*)d_in[3];
    const int*   hh_src    = (const int*)d_in[4];
    const int*   hh_dst    = (const int*)d_in[5];
    const int*   hd_src    = (const int*)d_in[6];
    const int*   hd_dst    = (const int*)d_in[7];
    const float* enc_src_w = (const float*)d_in[8];
    const float* enc_src_b = (const float*)d_in[9];
    const float* enc_dst_w = (const float*)d_in[10];
    const float* enc_dst_b = (const float*)d_in[11];
    const float* enc_wl    = (const float*)d_in[12];
    const float* enc_bl    = (const float*)d_in[13];
    const float* enc_wr    = (const float*)d_in[14];
    const float* proc_wl   = (const float*)d_in[15];
    const float* proc_bl   = (const float*)d_in[16];
    const float* proc_wr   = (const float*)d_in[17];
    const float* proc_mlp_w= (const float*)d_in[18];
    const float* proc_mlp_b= (const float*)d_in[19];
    const float* dec_src_w = (const float*)d_in[20];
    const float* dec_src_b = (const float*)d_in[21];
    const float* dec_dst_w = (const float*)d_in[22];
    const float* dec_dst_b = (const float*)d_in[23];
    const float* dec_wl    = (const float*)d_in[24];
    const float* dec_bl    = (const float*)d_in[25];
    const float* dec_wr    = (const float*)d_in[26];
    const float* out_w     = (const float*)d_in[27];
    const float* out_b     = (const float*)d_in[28];

    // -------- workspace layout --------
    const size_t SZD = (size_t)N_DATA * DIM * sizeof(float);
    const size_t SZH = (size_t)N_HID  * DIM * sizeof(float);
    char* p = (char*)d_ws;
    float* db0   = (float*)p; p += SZD;   // h_data / agg_data / final d
    float* db1   = (float*)p; p += SZD;   // dec dst activations
    float* hb0   = (float*)p; p += SZH;
    float* hb1   = (float*)p; p += SZH;
    float* hb2   = (float*)p; p += SZH;
    float* cnt_d = (float*)p; p += (size_t)N_DATA * sizeof(float);
    float* cnt_h = (float*)p; p += (size_t)N_HID  * sizeof(float);
    const size_t WN = (size_t)DIM * DIM;
    _Float16* w16_enc_wl  = (_Float16*)p; p += WN * 2;
    _Float16* w16_enc_wr  = (_Float16*)p; p += WN * 2;
    _Float16* w16_proc_wl = (_Float16*)p; p += WN * 2;
    _Float16* w16_proc_wr = (_Float16*)p; p += WN * 2;
    _Float16* w16_mlp_w   = (_Float16*)p; p += WN * 2;
    _Float16* w16_dec_sw  = (_Float16*)p; p += WN * 2;
    _Float16* w16_dec_wl  = (_Float16*)p; p += WN * 2;
    _Float16* w16_dec_wr  = (_Float16*)p; p += WN * 2;

    const int TB = 256;
    dim3 blk(TB);

    // -------- weight conversion to f16 --------
    {
        dim3 g((WN + TB - 1) / TB);
        cvt_f16_kernel<<<g, blk, 0, stream>>>(enc_wl,     w16_enc_wl,  (int)WN);
        cvt_f16_kernel<<<g, blk, 0, stream>>>(enc_wr,     w16_enc_wr,  (int)WN);
        cvt_f16_kernel<<<g, blk, 0, stream>>>(proc_wl,    w16_proc_wl, (int)WN);
        cvt_f16_kernel<<<g, blk, 0, stream>>>(proc_wr,    w16_proc_wr, (int)WN);
        cvt_f16_kernel<<<g, blk, 0, stream>>>(proc_mlp_w, w16_mlp_w,   (int)WN);
        cvt_f16_kernel<<<g, blk, 0, stream>>>(dec_src_w,  w16_dec_sw,  (int)WN);
        cvt_f16_kernel<<<g, blk, 0, stream>>>(dec_wl,     w16_dec_wl,  (int)WN);
        cvt_f16_kernel<<<g, blk, 0, stream>>>(dec_wr,     w16_dec_wr,  (int)WN);
    }

    // -------- encoder: node feature lifts --------
    enc_small_kernel<8><<<dim3((N_DATA * DIM) / TB), blk, 0, stream>>>(
        data_x, enc_src_w, enc_src_b, db0, N_DATA);
    enc_small_kernel<3><<<dim3((N_HID * DIM) / TB), blk, 0, stream>>>(
        hidden_x, enc_dst_w, enc_dst_b, hb0, N_HID);

    // -------- encoder SAGE: data -> hidden --------
    (void)hipMemsetAsync(hb1,   0, SZH, stream);
    (void)hipMemsetAsync(cnt_h, 0, (size_t)N_HID * sizeof(float), stream);
    scatter_add_kernel<<<dim3((E_CNT * 32) / TB), blk, 0, stream>>>(
        db0, dh_src, dh_dst, hb1, cnt_h, E_CNT);
    gemm256_wmma_kernel<false,false,false,false,true><<<dim3(N_HID / 64), blk, 0, stream>>>(
        hb1, w16_enc_wl, enc_bl, hb2, cnt_h, N_HID);          // hb2 = mean @ wl^T + bl
    gemm256_wmma_kernel<true,false,true,false,false><<<dim3(N_HID / 64), blk, 0, stream>>>(
        hb0, w16_enc_wr, nullptr, hb2, nullptr, N_HID);       // hb2 = gelu(hb2 + h_hid @ wr^T)

    // -------- processor SAGE: hidden -> hidden --------
    (void)hipMemsetAsync(hb1,   0, SZH, stream);
    (void)hipMemsetAsync(cnt_h, 0, (size_t)N_HID * sizeof(float), stream);
    scatter_add_kernel<<<dim3((E_CNT * 32) / TB), blk, 0, stream>>>(
        hb2, hh_src, hh_dst, hb1, cnt_h, E_CNT);
    gemm256_wmma_kernel<false,false,false,false,true><<<dim3(N_HID / 64), blk, 0, stream>>>(
        hb1, w16_proc_wl, proc_bl, hb0, cnt_h, N_HID);        // hb0 = mean @ wl^T + bl
    gemm256_wmma_kernel<true,false,false,false,false><<<dim3(N_HID / 64), blk, 0, stream>>>(
        hb2, w16_proc_wr, nullptr, hb0, nullptr, N_HID);      // hb0 += h_hid @ wr^T
    // MLP: hb1 = gelu( gelu(hb0) @ mlp_w^T + mlp_b )
    gemm256_wmma_kernel<false,true,true,false,false><<<dim3(N_HID / 64), blk, 0, stream>>>(
        hb0, w16_mlp_w, proc_mlp_b, hb1, nullptr, N_HID);

    // -------- decoder --------
    // s = gelu(h_hid @ dec_src_w^T + b) -> hb2
    gemm256_wmma_kernel<false,false,true,false,false><<<dim3(N_HID / 64), blk, 0, stream>>>(
        hb1, w16_dec_sw, dec_src_b, hb2, nullptr, N_HID);
    // d = gelu(data_x @ dec_dst_w^T + b) -> db1
    enc_small_kernel<8><<<dim3((N_DATA * DIM) / TB), blk, 0, stream>>>(
        data_x, dec_dst_w, dec_dst_b, db1, N_DATA);
    // SAGE hidden -> data
    (void)hipMemsetAsync(db0,   0, SZD, stream);
    (void)hipMemsetAsync(cnt_d, 0, (size_t)N_DATA * sizeof(float), stream);
    scatter_add_kernel<<<dim3((E_CNT * 32) / TB), blk, 0, stream>>>(
        hb2, hd_src, hd_dst, db0, cnt_d, E_CNT);
    // db0 = mean(db0) @ dec_wl^T + bl   (A aliases C; barrier-protected; fused mean)
    gemm256_wmma_kernel<false,false,false,true,true><<<dim3(N_DATA / 64), blk, 0, stream>>>(
        db0, w16_dec_wl, dec_bl, db0, cnt_d, N_DATA);
    // db0 = gelu(db0 + d @ dec_wr^T)
    gemm256_wmma_kernel<true,false,true,false,false><<<dim3(N_DATA / 64), blk, 0, stream>>>(
        db1, w16_dec_wr, nullptr, db0, nullptr, N_DATA);

    // -------- output projection --------
    out_proj_kernel<<<dim3((N_DATA * 8) / TB), blk, 0, stream>>>(
        db0, out_w, out_b, (float*)d_out, N_DATA);
}
